// MultiHeadAttention_68702296867518
// MI455X (gfx1250) — matrix operand order repaired and checked by reading, not compiled, not run on hardware
//
#include <hip/hip_runtime.h>
#include <stdint.h>

typedef unsigned short u16;
typedef __bf16 bf16_t;
typedef bf16_t v16bf __attribute__((ext_vector_type(16)));
typedef float  v8f   __attribute__((ext_vector_type(8)));

union FragBF {
  v16bf v;
  uint4 q[2];
  u16   h[16];
};

static constexpr int Bv    = 2;
static constexpr int Sv    = 2048;
static constexpr int Dv    = 1024;
static constexpr int Hv    = 16;
static constexpr int DKv   = 64;
static constexpr int Mrows = Bv * Sv;   // 4096

static constexpr int TM  = 128;         // block tile M
static constexpr int TNB = 128;         // block tile N
static constexpr int TK  = 32;          // K step
static constexpr int LDT = 40;          // padded LDS row stride (halves)

// ---------------- gfx1250 async global->LDS path (guarded) -----------------
#if __has_builtin(__builtin_amdgcn_global_load_async_to_lds_b128)
#define HAVE_ASYNC_LDS 1
typedef int v4i_vs __attribute__((vector_size(16)));
typedef __attribute__((address_space(1))) v4i_vs* gptr_v4i;  // __device__ int4*
typedef __attribute__((address_space(3))) v4i_vs* lptr_v4i;  // __shared__ int4*
#else
#define HAVE_ASYNC_LDS 0
#endif

__device__ __forceinline__ void copy_b128_to_lds(const u16* g, u16* l) {
#if HAVE_ASYNC_LDS
  __builtin_amdgcn_global_load_async_to_lds_b128((gptr_v4i)g, (lptr_v4i)l, 0, 0);
#else
  *(uint4*)l = *(const uint4*)g;   // sync fallback (loadcnt + ds_store)
#endif
}

__device__ __forceinline__ void wait_async_le4() {
#if HAVE_ASYNC_LDS
#if __has_builtin(__builtin_amdgcn_s_wait_asynccnt)
  __builtin_amdgcn_s_wait_asynccnt(4);
#else
  asm volatile("s_wait_asynccnt 0x4" ::: "memory");
#endif
#endif
}

__device__ __forceinline__ void wait_async_0() {
#if HAVE_ASYNC_LDS
#if __has_builtin(__builtin_amdgcn_s_wait_asynccnt)
  __builtin_amdgcn_s_wait_asynccnt(0);
#else
  asm volatile("s_wait_asynccnt 0x0" ::: "memory");
#endif
#endif
}

__device__ __forceinline__ u16 f2bf(float f) {
  union { float f; unsigned int u; } x; x.f = f;
  unsigned int r = x.u + 0x7FFFu + ((x.u >> 16) & 1u);  // round-to-nearest-even
  return (u16)(r >> 16);
}

__device__ __forceinline__ v8f wmma_bf16(v16bf a, v16bf b, v8f c) {
  // D = A(16x32 bf16) * B(32x16 bf16) + C(16x16 f32)
  return __builtin_amdgcn_wmma_f32_16x16x32_bf16(false, a, false, b, (short)0, c,
                                                 false, false);
}

// ---------------- fp32 -> bf16 conversion (4 elems / thread) ----------------
__global__ __launch_bounds__(256) void cvt_f32_bf16(const float* __restrict__ src,
                                                    u16* __restrict__ dst, int n) {
  int idx = (blockIdx.x * 256 + threadIdx.x) * 4;
  if (idx >= n) return;
  float4 f = *(const float4*)(src + idx);
  union { u16 u[4]; uint2 v; } r;
  r.u[0] = f2bf(f.x); r.u[1] = f2bf(f.y); r.u[2] = f2bf(f.z); r.u[3] = f2bf(f.w);
  *(uint2*)(dst + idx) = r.v;
}

// ---------------- NT GEMM: C[M,N] = A[M,K] * B[N,K]^T  (x @ W.T) ------------
// 256 threads = 8 waves; block tile 128x128; K-step 32; async double-buffered
// LDS staging. Wave grid 4(M) x 2(N): each wave owns 32x64 = 2x4 WMMA tiles.
__device__ __forceinline__ void stage_tiles(const u16* __restrict__ A,
                                            const u16* __restrict__ Bw,
                                            int K, int m0, int n0, int k0,
                                            u16* as, u16* bs, int tid) {
#pragma unroll
  for (int i = 0; i < 2; ++i) {          // A tile: 128 rows x 32 halves
    int idx = tid + i * 256;             // 512 uint4 segments
    int r = idx >> 2, s = idx & 3;
    copy_b128_to_lds(&A[(size_t)(m0 + r) * K + k0 + s * 8], &as[r * LDT + s * 8]);
  }
#pragma unroll
  for (int i = 0; i < 2; ++i) {          // B tile: 128 rows x 32 halves
    int idx = tid + i * 256;
    int r = idx >> 2, s = idx & 3;
    copy_b128_to_lds(&Bw[(size_t)(n0 + r) * K + k0 + s * 8], &bs[r * LDT + s * 8]);
  }
}

__global__ __launch_bounds__(256) void gemm_nt(const u16* __restrict__ A,
                                               const u16* __restrict__ Bw,
                                               void* __restrict__ Cout,
                                               int M, int N, int K, int out_bf16) {
  __shared__ u16 As[2][TM * LDT];
  __shared__ u16 Bs[2][TNB * LDT];
  const int tid  = threadIdx.x;
  const int lane = tid & 31;
  const int wave = tid >> 5;
  const int wm   = (wave & 3) * 32;   // 4 waves along M
  const int wn   = (wave >> 2) * 64;  // 2 waves along N
  const int m0   = blockIdx.y * TM;
  const int n0   = blockIdx.x * TNB;
  const int g    = lane >> 4;         // lane group (0: lanes 0-15, 1: 16-31)
  const int c    = lane & 15;

  v8f acc[2][4] = {};
  const int nk = K / TK;

  stage_tiles(A, Bw, K, m0, n0, 0, As[0], Bs[0], tid);

  for (int kt = 0; kt < nk; ++kt) {
    const int cur = kt & 1;
    if (kt + 1 < nk) {
      stage_tiles(A, Bw, K, m0, n0, (kt + 1) * TK, As[cur ^ 1], Bs[cur ^ 1], tid);
      wait_async_le4();    // previous batch (4 ops) has landed
    } else {
      wait_async_0();
    }
    __syncthreads();

    const u16* as = As[cur];
    const u16* bs = Bs[cur];
    // A fragments: lane holds row m=c, contiguous K chunk
    // [16g, 16g+16).
    FragBF a[2], bf[4];
#pragma unroll
    for (int i = 0; i < 2; ++i) {
      const u16* ar = &as[(wm + i * 16 + c) * LDT];
      a[i].q[0] = *(const uint4*)(ar + 16 * g);
      a[i].q[1] = *(const uint4*)(ar + 16 * g + 8);
    }
    // B fragments: lane holds column n=c of Bwmma (= row n of B), contiguous
    // K chunk [16g, 16g+16).
#pragma unroll
    for (int j = 0; j < 4; ++j) {
      const u16* br = &bs[(wn + j * 16 + c) * LDT + 16 * g];
      bf[j].q[0] = *(const uint4*)(br);
      bf[j].q[1] = *(const uint4*)(br + 8);
    }
#pragma unroll
    for (int i = 0; i < 2; ++i)
#pragma unroll
      for (int j = 0; j < 4; ++j)
        acc[i][j] = wmma_bf16(a[i].v, bf[j].v, acc[i][j]);
    __syncthreads();
  }

  // D layout: vgpr ii of lane -> row ii + 8g, col c.
#pragma unroll
  for (int i = 0; i < 2; ++i)
#pragma unroll
    for (int j = 0; j < 4; ++j)
#pragma unroll
      for (int ii = 0; ii < 8; ++ii) {
        int row = m0 + wm + i * 16 + ii + 8 * g;
        int col = n0 + wn + j * 16 + c;
        float v = acc[i][j][ii];
        if (out_bf16)
          ((u16*)Cout)[(size_t)row * N + col] = f2bf(v);
        else
          ((float*)Cout)[(size_t)row * N + col] = v;
      }
}

// ---------------- Flash attention: one wave per (b, h, 16-query tile) ------
// q/k/v are bf16 [B, S, H*DK] row-major; head h occupies cols [64h, 64h+64).
__global__ __launch_bounds__(32) void flash_attn(const u16* __restrict__ qp,
                                                 const u16* __restrict__ kp,
                                                 const u16* __restrict__ vp,
                                                 u16* __restrict__ ctx) {
  __shared__ u16 Plds[16 * LDT];
  const int lane = threadIdx.x;
  const int qt   = blockIdx.x;    // query tile (16 rows)
  const int h    = blockIdx.y;
  const int b    = blockIdx.z;
  const int g    = lane >> 4;
  const int c    = lane & 15;
  const size_t base = ((size_t)b * Sv) * Dv + (size_t)h * DKv;

  // Q fragments stay in registers: A layout, K=64 split into two K=32 chunks.
  FragBF qa[2];
  {
    const u16* qr = qp + base + (size_t)(qt * 16 + c) * Dv;
#pragma unroll
    for (int c2 = 0; c2 < 2; ++c2) {
      qa[c2].q[0] = *(const uint4*)(qr + c2 * 32 + 16 * g);
      qa[c2].q[1] = *(const uint4*)(qr + c2 * 32 + 16 * g + 8);
    }
  }

  float m_i[8], l_i[8];
  v8f o[4] = {};
#pragma unroll
  for (int i = 0; i < 8; ++i) { m_i[i] = -3.0e38f; l_i[i] = 0.0f; }

  const int kend = qt * 16 + 16;  // causal: keys [0, kend)
  for (int j0 = 0; j0 < kend; j0 += 32) {
    // Scores S[16q, 32k] = two 16x16 WMMA tiles, each accumulating K=64.
    v8f s[2] = {};
#pragma unroll
    for (int n = 0; n < 2; ++n) {
      const u16* kr = kp + base + (size_t)(j0 + 16 * n + c) * Dv;  // key row
#pragma unroll
      for (int c2 = 0; c2 < 2; ++c2) {
        FragBF kb;  // B frag: lane = column (key), contiguous dims [16g,16g+16)
        kb.q[0] = *(const uint4*)(kr + c2 * 32 + 16 * g);
        kb.q[1] = *(const uint4*)(kr + c2 * 32 + 16 * g + 8);
        s[n] = wmma_bf16(qa[c2].v, kb.v, s[n]);
      }
    }

    // Mask-before-scale (reference order), then online softmax.
#pragma unroll
    for (int i = 0; i < 8; ++i) {
      int q_abs = qt * 16 + i + 8 * g;          // row for vgpr i (D layout)
      float s0 = (j0 + c      <= q_abs) ? s[0][i] * 0.125f : -1.0e30f;
      float s1 = (j0 + 16 + c <= q_abs) ? s[1][i] * 0.125f : -1.0e30f;
      float rmax = fmaxf(s0, s1);
#pragma unroll
      for (int off = 8; off >= 1; off >>= 1)     // reduce across 16 cols
        rmax = fmaxf(rmax, __shfl_xor(rmax, off, 32));
      float mnew  = fmaxf(m_i[i], rmax);
      float alpha = __expf(m_i[i] - mnew);
      float p0 = __expf(s0 - mnew);
      float p1 = __expf(s1 - mnew);
      float rsum = p0 + p1;
#pragma unroll
      for (int off = 8; off >= 1; off >>= 1)
        rsum += __shfl_xor(rsum, off, 32);
      l_i[i] = l_i[i] * alpha + rsum;
      m_i[i] = mnew;
#pragma unroll
      for (int t = 0; t < 4; ++t) o[t][i] *= alpha;  // rescale O rows
      int r = i + 8 * g;
      Plds[r * LDT + c]      = f2bf(p0);   // transpose P via LDS
      Plds[r * LDT + 16 + c] = f2bf(p1);
    }
    __syncthreads();  // single wave: NOP; LDS is in-order per wave

    FragBF pa;  // P in A layout (16x32, K = 32 local keys)
    {
      const u16* pr = &Plds[c * LDT];
      pa.q[0] = *(const uint4*)(pr + 16 * g);
      pa.q[1] = *(const uint4*)(pr + 16 * g + 8);
    }
    // O[16,64] += P[16,32] * V[32,64]  (4 N-tiles of 16 dims)
#pragma unroll
    for (int t = 0; t < 4; ++t) {
      FragBF vb;  // B frag of V: column = dim 16t+c, K = keys [j0+16g, +16)
      const u16* vc = vp + base + 16 * t + c;
#pragma unroll
      for (int kk = 0; kk < 16; ++kk)
        vb.h[kk] = vc[(size_t)(j0 + 16 * g + kk) * Dv];
      o[t] = wmma_bf16(pa.v, vb.v, o[t]);
    }
    __syncthreads();
  }

  // Normalize and store context in [B, S, H*DK] layout (bf16).
#pragma unroll
  for (int t = 0; t < 4; ++t) {
#pragma unroll
    for (int i = 0; i < 8; ++i) {
      float inv = 1.0f / l_i[i];
      int row = qt * 16 + i + 8 * g;
      ctx[base + (size_t)row * Dv + 16 * t + c] = f2bf(o[t][i] * inv);
    }
  }
}

// ---------------------------------------------------------------------------
extern "C" void kernel_launch(void* const* d_in, const int* in_sizes, int n_in,
                              void* d_out, int out_size, void* d_ws, size_t ws_size,
                              hipStream_t stream) {
  const float* Q  = (const float*)d_in[0];
  const float* K  = (const float*)d_in[1];
  const float* V  = (const float*)d_in[2];
  const float* Wq = (const float*)d_in[3];
  const float* Wk = (const float*)d_in[4];
  const float* Wv = (const float*)d_in[5];
  const float* Wo = (const float*)d_in[6];
  // d_in[7] = causal mask: recomputed analytically, not read.
  float* out = (float*)d_out;

  char* ws = (char*)d_ws;
  const size_t SZ_ACT = (size_t)Mrows * Dv * sizeof(u16);  // 8 MiB
  const size_t SZ_W   = (size_t)Dv * Dv * sizeof(u16);     // 2 MiB
  u16* Qb  = (u16*)(ws + 0 * SZ_ACT);
  u16* Kb  = (u16*)(ws + 1 * SZ_ACT);
  u16* Vb  = (u16*)(ws + 2 * SZ_ACT);
  u16* Wqb = (u16*)(ws + 3 * SZ_ACT + 0 * SZ_W);
  u16* Wkb = (u16*)(ws + 3 * SZ_ACT + 1 * SZ_W);
  u16* Wvb = (u16*)(ws + 3 * SZ_ACT + 2 * SZ_W);
  u16* Wob = (u16*)(ws + 3 * SZ_ACT + 3 * SZ_W);
  u16* qp  = (u16*)(ws + 3 * SZ_ACT + 4 * SZ_W);
  u16* kp  = (u16*)(ws + 4 * SZ_ACT + 4 * SZ_W);
  u16* vp  = (u16*)(ws + 5 * SZ_ACT + 4 * SZ_W);
  u16* ctx = (u16*)(ws + 6 * SZ_ACT + 4 * SZ_W);
  (void)ws_size; (void)in_sizes; (void)n_in; (void)out_size;

  const int nAct = Mrows * Dv;  // 4,194,304
  const int nW   = Dv * Dv;     // 1,048,576
  cvt_f32_bf16<<<nAct / (4 * 256), 256, 0, stream>>>(Q, Qb, nAct);
  cvt_f32_bf16<<<nAct / (4 * 256), 256, 0, stream>>>(K, Kb, nAct);
  cvt_f32_bf16<<<nAct / (4 * 256), 256, 0, stream>>>(V, Vb, nAct);
  cvt_f32_bf16<<<nW / (4 * 256), 256, 0, stream>>>(Wq, Wqb, nW);
  cvt_f32_bf16<<<nW / (4 * 256), 256, 0, stream>>>(Wk, Wkb, nW);
  cvt_f32_bf16<<<nW / (4 * 256), 256, 0, stream>>>(Wv, Wvb, nW);
  cvt_f32_bf16<<<nW / (4 * 256), 256, 0, stream>>>(Wo, Wob, nW);

  dim3 ggrid(Dv / TNB, Mrows / TM);  // (8, 32)
  gemm_nt<<<ggrid, 256, 0, stream>>>(Qb, Wqb, qp, Mrows, Dv, Dv, 1);
  gemm_nt<<<ggrid, 256, 0, stream>>>(Kb, Wkb, kp, Mrows, Dv, Dv, 1);
  gemm_nt<<<ggrid, 256, 0, stream>>>(Vb, Wvb, vp, Mrows, Dv, Dv, 1);

  flash_attn<<<dim3(Sv / 16, Hv, Bv), 32, 0, stream>>>(qp, kp, vp, ctx);

  gemm_nt<<<ggrid, 256, 0, stream>>>(ctx, Wob, out, Mrows, Dv, Dv, 0);
}
